// LocalNeighbourEmbedding_65335042506818
// MI455X (gfx1250) — matrix operand order repaired and checked by reading, not compiled, not run on hardware
//
#include <hip/hip_runtime.h>
#include <hip/hip_bf16.h>
#include <math.h>

typedef __attribute__((ext_vector_type(16))) __bf16 v16bf;
typedef __attribute__((ext_vector_type(8)))  __bf16 v8bf;
typedef __attribute__((ext_vector_type(2)))  __bf16 v2bf;
typedef __attribute__((ext_vector_type(8)))  float  v8f;

#define NRES   65536
#define KNB    16
#define FSTR   168   // padded feature K-stride (131 used -> 160 compute, 168 for bank spread)
#define HID    256
#define HSTR   264   // padded hidden K-stride
#define OUTD   128
#define NWAVES 8
#define BLOCK  256
#define GRID   512

// gelu (tanh approximation, matches jax.nn.gelu approximate=True)
__device__ __forceinline__ float gelu_tanh(float x) {
  const float c0 = 0.7978845608028654f;  // sqrt(2/pi)
  const float c1 = 0.044715f;
  float y = c0 * (x + c1 * x * x * x);
#if __has_builtin(__builtin_amdgcn_tanhf)
  return 0.5f * x * (1.0f + __builtin_amdgcn_tanhf(y));
#else
  return x / (1.0f + __expf(-2.0f * y));
#endif
}

__device__ __forceinline__ v16bf cat8(v8bf lo, v8bf hi) {
  return __builtin_shufflevector(lo, hi, 0,1,2,3,4,5,6,7,8,9,10,11,12,13,14,15);
}

__global__ __launch_bounds__(BLOCK)
void local_nbr_embed_kernel(const float* __restrict__ ca,
                            const int*   __restrict__ resi,
                            const int*   __restrict__ chain,
                            const int*   __restrict__ nbr,
                            const float* __restrict__ w1,
                            const float* __restrict__ b1,
                            const float* __restrict__ w2,
                            const float* __restrict__ b2,
                            const float* __restrict__ gamma,
                            const float* __restrict__ beta,
                            float* __restrict__ out)
{
  __shared__ __bf16 sW1[HID * FSTR];          // w1^T : [n][k]; k=130 carries b1[n]
  __shared__ __bf16 sW2[OUTD * HSTR];         // w2^T : [n][k], k zero-padded 256->264
  __shared__ __bf16 sFeat[NWAVES][16 * FSTR]; // per-wave 16x160 feature tile (A of GEMM1)
  __shared__ __bf16 sH[NWAVES][16 * HSTR];    // per-wave 16x256 hidden tile (A of GEMM2)
  __shared__ float  sMsg[NWAVES][OUTD];
  __shared__ float  sB2[OUTD];

  const int tid = threadIdx.x;

  // ---- stage weights once per block: fp32 global -> bf16 LDS, transposed.
  //      w1^T column k=130 holds b1 (pairs with constant-1 feature column). ----
  for (int n = tid; n < HID; n += BLOCK)
    for (int k = 0; k < FSTR; ++k)
      sW1[n * FSTR + k] = (k < 130) ? (__bf16)w1[k * HID + n]
                        : (k == 130) ? (__bf16)b1[n] : (__bf16)0.0f;
  for (int n = tid; n < OUTD; n += BLOCK)
    for (int k = 0; k < HSTR; ++k)
      sW2[n * HSTR + k] = (k < HID) ? (__bf16)w2[k * OUTD + n] : (__bf16)0.0f;
  for (int i = tid; i < OUTD; i += BLOCK) sB2[i] = b2[i];

  const int wv     = tid >> 5;
  const int lane   = tid & 31;
  const int e      = lane & 15;        // edge / matrix row or column index
  const int half   = lane >> 4;
  const int mbase  = half ? 8 : 0;     // D-layout: rows held by this lane start here
  const int kApart = half ? 8 : 0;     // A-fragment K sub-offset per ISA layout
  const int kBpart = half ? 8 : 0;     // B-fragment K sub-offset per ISA layout

  // ---- zero this wave's feature tile once, then plant the constant-1 bias
  //      column (k=130). Per-iteration writes only touch k<64 and k=64+rd<=129,
  //      so both the zeros and the 1.0 persist across residues. ----
  {
    v8bf z = {};
    for (int i = lane; i < (16 * FSTR) / 8; i += 32)
      *(v8bf*)(&sFeat[wv][i * 8]) = z;
  }
  __builtin_amdgcn_wave_barrier();
  if (half == 0) sFeat[wv][e * FSTR + 130] = (__bf16)1.0f;
  __syncthreads();

  __bf16* frow = &sFeat[wv][e * FSTR];
  const int totalWaves = GRID * NWAVES;

  for (int r = blockIdx.x * NWAVES + wv; r < NRES; r += totalWaves) {
    // ---------- per-edge geometry (each edge computed by both lane halves) ----------
    int  j     = nbr[r * KNB + e];
    bool valid = (j >= 0);
    int  jj    = valid ? j : (NRES - 1);          // -1 wraps like jnp gather; masked later
    float dx = ca[r*3+0] - ca[jj*3+0];
    float dy = ca[r*3+1] - ca[jj*3+1];
    float dz = ca[r*3+2] - ca[jj*3+2];
    float dist = sqrtf(dx*dx + dy*dy + dz*dz);
    int rd = resi[r] - resi[jj];
    rd = min(max(rd, -32), 32) + 32;
    if (chain[r] != chain[jj]) rd = 65;
    unsigned mask = __builtin_amdgcn_ballot_w32(valid) & 0xFFFFu;
    float cnt = (float)__popc(mask);              // # valid edges (for b2 folding)
    float rowm[8];
    #pragma unroll
    for (int v = 0; v < 8; ++v)
      rowm[v] = ((mask >> (mbase + v)) & 1u) ? 1.0f : 0.0f;

    // ---------- features -> LDS: lower half writes RBF, upper half sets one-hot ----------
    if (half == 0) {
      // (dist - c_k)/sigma = dist*(64/22) - k*(64/63)
      float dscaled = dist * (64.0f / 22.0f);
      #pragma unroll
      for (int k = 0; k < 64; k += 2) {          // RBF bins, centers linspace(0,22,64)
        float t0 = dscaled - (float)k     * (64.0f / 63.0f);
        float t1 = dscaled - (float)(k+1) * (64.0f / 63.0f);
        v2bf p; p.x = (__bf16)__expf(-t0*t0); p.y = (__bf16)__expf(-t1*t1);
        *(v2bf*)(frow + k) = p;
      }
    } else {
      frow[64 + rd] = (__bf16)1.0f;              // one-hot set (rest stays zero)
    }
    __builtin_amdgcn_wave_barrier();   // wave-local LDS RAW: in-order DS + sched fence

    // ---------- GEMM1 (+bias): feat[16x160] @ w1[160x256] -> h ----------
    v16bf afr[5];
    const __bf16* ap = &sFeat[wv][e * FSTR + kApart];
    #pragma unroll
    for (int ks = 0; ks < 5; ++ks) {
      v8bf lo = *(const v8bf*)(ap + ks*32);
      v8bf hi = *(const v8bf*)(ap + ks*32 + 16);
      afr[ks] = cat8(lo, hi);
    }
    // one-hot reset: LDS ops are in-order within the wave, so this store
    // completes after the fragment loads above; tile is clean for next iter.
    if (half) frow[64 + rd] = (__bf16)0.0f;

    #pragma unroll
    for (int nt = 0; nt < 16; ++nt) {
      int n = nt * 16 + e;
      const __bf16* bp = &sW1[n * FSTR + kBpart];
      v8f acc = {0.f,0.f,0.f,0.f,0.f,0.f,0.f,0.f};
      #pragma unroll
      for (int ks = 0; ks < 5; ++ks) {
        v16bf bb = cat8(*(const v8bf*)(bp + ks*32), *(const v8bf*)(bp + ks*32 + 16));
        acc = __builtin_amdgcn_wmma_f32_16x16x32_bf16(false, afr[ks], false, bb,
                                                      (short)0, acc, false, false);
      }
      #pragma unroll
      for (int v = 0; v < 8; ++v) {
        float g = gelu_tanh(acc[v]);             // bias already accumulated via k=130
        sH[wv][(mbase + v) * HSTR + n] = (__bf16)g;
      }
    }
    __builtin_amdgcn_wave_barrier();

    // ---------- GEMM2: h[16x256] @ w2[256x128], masked mean over 16 edges ----------
    v16bf hfr[8];
    const __bf16* hp = &sH[wv][e * HSTR + kApart];
    #pragma unroll
    for (int ks = 0; ks < 8; ++ks) {
      v8bf lo = *(const v8bf*)(hp + ks*32);
      v8bf hi = *(const v8bf*)(hp + ks*32 + 16);
      hfr[ks] = cat8(lo, hi);
    }
    #pragma unroll
    for (int nt = 0; nt < 8; ++nt) {
      int n = nt * 16 + e;
      const __bf16* bp = &sW2[n * HSTR + kBpart];
      v8f acc = {0.f,0.f,0.f,0.f,0.f,0.f,0.f,0.f};
      #pragma unroll
      for (int ks = 0; ks < 8; ++ks) {
        v16bf bb = cat8(*(const v8bf*)(bp + ks*32), *(const v8bf*)(bp + ks*32 + 16));
        acc = __builtin_amdgcn_wmma_f32_16x16x32_bf16(false, hfr[ks], false, bb,
                                                      (short)0, acc, false, false);
      }
      float s = 0.f;
      #pragma unroll
      for (int v = 0; v < 8; ++v)
        s = fmaf(rowm[v], acc[v], s);            // masked row sum, b2 folded below
      s += __shfl_down(s, 16, 32);               // fold upper-half rows into lanes 0..15
      if (lane < 16) sMsg[wv][n] = (s + cnt * sB2[n]) * (1.0f / 16.0f);
    }
    __builtin_amdgcn_wave_barrier();

    // ---------- LayerNorm over 128 channels + store ----------
    int c0 = lane * 4;
    float x0 = sMsg[wv][c0+0], x1 = sMsg[wv][c0+1];
    float x2 = sMsg[wv][c0+2], x3 = sMsg[wv][c0+3];
    float s1 = x0 + x1 + x2 + x3;
    float s2 = x0*x0 + x1*x1 + x2*x2 + x3*x3;
    #pragma unroll
    for (int off = 16; off > 0; off >>= 1) {
      s1 += __shfl_xor(s1, off, 32);
      s2 += __shfl_xor(s2, off, 32);
    }
    float mean = s1 * (1.0f / 128.0f);
    float var  = s2 * (1.0f / 128.0f) - mean * mean;
    float inv  = rsqrtf(var + 1e-5f);
    float4 o;
    o.x = (x0 - mean) * inv * gamma[c0+0] + beta[c0+0];
    o.y = (x1 - mean) * inv * gamma[c0+1] + beta[c0+1];
    o.z = (x2 - mean) * inv * gamma[c0+2] + beta[c0+2];
    o.w = (x3 - mean) * inv * gamma[c0+3] + beta[c0+3];
    *(float4*)(out + (size_t)r * OUTD + c0) = o;
  }
}

extern "C" void kernel_launch(void* const* d_in, const int* in_sizes, int n_in,
                              void* d_out, int out_size, void* d_ws, size_t ws_size,
                              hipStream_t stream) {
  const float* ca    = (const float*)d_in[0];
  const int*   resi  = (const int*)  d_in[1];
  const int*   chain = (const int*)  d_in[2];
  const int*   nbr   = (const int*)  d_in[3];
  const float* w1    = (const float*)d_in[4];
  const float* b1    = (const float*)d_in[5];
  const float* w2    = (const float*)d_in[6];
  const float* b2    = (const float*)d_in[7];
  const float* gamma = (const float*)d_in[8];
  const float* beta  = (const float*)d_in[9];
  float* out = (float*)d_out;
  local_nbr_embed_kernel<<<GRID, BLOCK, 0, stream>>>(
      ca, resi, chain, nbr, w1, b1, w2, b2, gamma, beta, out);
}
